// CausalAttention_72335839199357
// MI455X (gfx1250) — matrix operand order repaired and checked by reading, not compiled, not run on hardware
//
#include <hip/hip_runtime.h>
#include <hip/hip_bf16.h>

// ---------------------------------------------------------------------------
// Causal attention for MI455X (gfx1250, wave32, WMMA + TDM).
//   B=4, T=2048, D=1024, fp32 in/out.  Compute-bound => bf16 WMMA everywhere.
//   ws layout: Q bf16 [B*T, D] | K bf16 [B*T, D] | V^T bf16 [B, D, T]
// ---------------------------------------------------------------------------

typedef __bf16 bf16x16 __attribute__((ext_vector_type(16)));
typedef __bf16 bf16x8  __attribute__((ext_vector_type(8)));
typedef __bf16 bf16x4  __attribute__((ext_vector_type(4)));
typedef float  f32x8   __attribute__((ext_vector_type(8)));
typedef unsigned int u32x4 __attribute__((ext_vector_type(4)));
typedef int    i32x8   __attribute__((ext_vector_type(8)));
typedef int    i32x4   __attribute__((ext_vector_type(4)));

union BFrag { bf16x16 v; bf16x8 h[2]; };

#define WMMA_BF16(a, b, c) \
    __builtin_amdgcn_wmma_f32_16x16x32_bf16(false, (a), false, (b), (short)0, (c), false, false)

static constexpr int Bb = 4;
static constexpr int Tt = 2048;
static constexpr int Dd = 1024;
static constexpr int Mr = Bb * Tt; // 8192 rows of X

#if defined(__AMDGCN__) && __has_builtin(__builtin_amdgcn_tensor_load_to_lds) && \
    __has_builtin(__builtin_amdgcn_s_wait_tensorcnt)
#define HAVE_TDM 1
#else
#define HAVE_TDM 0
#endif

#if HAVE_TDM
// Issue a TDM DMA of a 2D bf16 tile [64 rows x 256 cols], row pitch 1024
// elements, from global `gsrc` into LDS at byte offset `lds_off`.
// Descriptor layout per CDNA5 ISA (D# groups 0/1).
// This toolchain's builtin takes 6 args (4 SGPR groups + spare group + cpol).
__device__ __forceinline__ void tdm_load_k_chunk(const __bf16* gsrc, unsigned lds_off)
{
    unsigned long long ga = (unsigned long long)(size_t)gsrc;
    u32x4 g0;
    g0.x = 1u;                                        // count=1, user descriptor
    g0.y = lds_off;                                   // lds_addr (bytes)
    g0.z = (unsigned)(ga & 0xffffffffull);            // global_addr[31:0]
    g0.w = (unsigned)((ga >> 32) & 0x1ffffffull)      // global_addr[56:32]
         | (2u << 30);                                // type=2 ("image")
    i32x8 g1;
    g1[0] = (int)(1u << 16);                          // data_size=1 (2B), mask=0
    g1[1] = (int)(1024u << 16);                       // tensor_dim0 = 1024
    g1[2] = (int)(2048u << 16);                       // tensor_dim1 = 2048
    g1[3] = (int)(256u  << 16);                       // tile_dim0 = 256
    g1[4] = 64;                                       // tile_dim1 = 64, tile_dim2 = 0
    g1[5] = 1024;                                     // tensor_dim0_stride = 1024
    g1[6] = 0;
    g1[7] = 0;
    i32x4 gz4 = {0, 0, 0, 0};                         // groups 2/3: unused (<=2D)
    i32x8 gz8 = {0, 0, 0, 0, 0, 0, 0, 0};
    __builtin_amdgcn_tensor_load_to_lds(g0, g1, gz4, gz4, gz8, 0);
}
#endif

// ---------------------------------------------------------------------------
// Kernel 1: Y = X @ W^T (torch Linear), fp32 in, bf16 out.
//   Block tile 128(M) x 64(N); 8 waves as 4(M) x 2(N); wave tile 32x32.
//   KT=64 staged in LDS as bf16 (load-all then convert-all so the global
//   loads issue as one clause).  transposedV: store as [B][D][T].
// ---------------------------------------------------------------------------
__global__ __launch_bounds__(256) void qkv_gemm_bf16(
    const float* __restrict__ X, const float* __restrict__ W,
    unsigned short* __restrict__ Yraw, int transposedV)
{
    __bf16* __restrict__ Y = reinterpret_cast<__bf16*>(Yraw);

    __shared__ __attribute__((aligned(16))) __bf16 Xs[128 * 64];
    __shared__ __attribute__((aligned(16))) __bf16 Ws[64 * 64];

    const int tid  = threadIdx.x;
    const int lane = tid & 31;
    const int w    = tid >> 5;
    const int wm   = w >> 1;  // 0..3
    const int wn   = w & 1;   // 0..1
    const int row0 = blockIdx.y * 128;
    const int col0 = blockIdx.x * 64;

    const int ln15 = lane & 15;
    const int kbA  = (lane >> 4) * 8;   // A fragment K-base
    const int kbB  = (lane >> 4) * 16;  // unused (B fragment also uses kbA)
    const int mhi  = (lane >> 4) * 8;   // C fragment M offset

    // Per-thread staging coordinates (fixed across K-steps).
    int xr[8], xc[8], wr[4], wc[4];
#pragma unroll
    for (int seg = 0; seg < 8; ++seg) {
        int e = (seg * 256 + tid) * 4;
        xr[seg] = e >> 6; xc[seg] = e & 63;
    }
#pragma unroll
    for (int seg = 0; seg < 4; ++seg) {
        int e = (seg * 256 + tid) * 4;
        wr[seg] = e >> 6; wc[seg] = e & 63;
    }

    const f32x8 z8 = {0.f, 0.f, 0.f, 0.f, 0.f, 0.f, 0.f, 0.f};
    f32x8 acc[2][2];
#pragma unroll
    for (int i = 0; i < 2; ++i)
#pragma unroll
        for (int j = 0; j < 2; ++j) acc[i][j] = z8;

    for (int k0 = 0; k0 < Dd; k0 += 64) {
        // --- phase 1: issue ALL staging loads (distinct temporaries)
        float4 xv[8], wv[4];
#pragma unroll
        for (int seg = 0; seg < 8; ++seg)
            xv[seg] = *(const float4*)(X + (size_t)(row0 + xr[seg]) * Dd + k0 + xc[seg]);
#pragma unroll
        for (int seg = 0; seg < 4; ++seg)
            wv[seg] = *(const float4*)(W + (size_t)(col0 + wr[seg]) * Dd + k0 + wc[seg]);

        // near-scope prefetch of next K-step tile (WGP-level, ISA §7.3)
        if (k0 + 64 < Dd) {
            __builtin_prefetch(X + (size_t)(row0 + xr[0]) * Dd + k0 + 64 + xc[0], 0, 3);
            __builtin_prefetch(W + (size_t)(col0 + wr[0]) * Dd + k0 + 64 + wc[0], 0, 3);
        }

        // --- phase 2: convert fp32 -> packed bf16, store to LDS
#pragma unroll
        for (int seg = 0; seg < 8; ++seg) {
            bf16x4 pk;
            pk[0] = (__bf16)xv[seg].x; pk[1] = (__bf16)xv[seg].y;
            pk[2] = (__bf16)xv[seg].z; pk[3] = (__bf16)xv[seg].w;
            *(bf16x4*)&Xs[xr[seg] * 64 + xc[seg]] = pk;
        }
#pragma unroll
        for (int seg = 0; seg < 4; ++seg) {
            bf16x4 pk;
            pk[0] = (__bf16)wv[seg].x; pk[1] = (__bf16)wv[seg].y;
            pk[2] = (__bf16)wv[seg].z; pk[3] = (__bf16)wv[seg].w;
            *(bf16x4*)&Ws[wr[seg] * 64 + wc[seg]] = pk;
        }
        __syncthreads();

#pragma unroll
        for (int ks = 0; ks < 2; ++ks) {
            BFrag a[2], b[2];
#pragma unroll
            for (int mt = 0; mt < 2; ++mt) {
                const __bf16* p = &Xs[(wm * 32 + mt * 16 + ln15) * 64 + ks * 32];
                a[mt].h[0] = *(const bf16x8*)(p + kbA);
                a[mt].h[1] = *(const bf16x8*)(p + kbA + 16);
            }
#pragma unroll
            for (int nt = 0; nt < 2; ++nt) {
                const __bf16* p = &Ws[(wn * 32 + nt * 16 + ln15) * 64 + ks * 32];
                b[nt].h[0] = *(const bf16x8*)(p + kbA);
                b[nt].h[1] = *(const bf16x8*)(p + kbA + 16);
            }
#pragma unroll
            for (int mt = 0; mt < 2; ++mt)
#pragma unroll
                for (int nt = 0; nt < 2; ++nt)
                    acc[mt][nt] = WMMA_BF16(a[mt].v, b[nt].v, acc[mt][nt]);
        }
        __syncthreads();
    }

    // --- store bf16 (optionally transposed per batch: [B][D][T])
#pragma unroll
    for (int mt = 0; mt < 2; ++mt) {
#pragma unroll
        for (int nt = 0; nt < 2; ++nt) {
#pragma unroll
            for (int r = 0; r < 8; ++r) {
                int gm = row0 + wm * 32 + mt * 16 + mhi + r;
                int gn = col0 + wn * 32 + nt * 16 + ln15;
                __bf16 val = (__bf16)acc[mt][nt][r];
                if (!transposedV) {
                    Y[(size_t)gm * Dd + gn] = val;
                } else {
                    int bb = gm >> 11;
                    int t  = gm & (Tt - 1);
                    Y[(size_t)bb * Dd * Tt + (size_t)gn * Tt + t] = val;
                }
            }
        }
    }
}

// ---------------------------------------------------------------------------
// Kernel 2: causal flash attention, single "head" of width D=1024.
//   One block per (batch, 32-query tile); 256 threads = 8 waves.
//   K tile streamed into LDS in 64x256 chunks by the Tensor Data Mover,
//   double-buffered so DMA overlaps WMMA.  V read via transposed global.
// ---------------------------------------------------------------------------
__global__ __launch_bounds__(256) void flash_attn_bf16(
    const unsigned short* __restrict__ Qraw,
    const unsigned short* __restrict__ Kraw,
    const unsigned short* __restrict__ Vtraw,
    float* __restrict__ Out)
{
    const __bf16* __restrict__ Qg  = reinterpret_cast<const __bf16*>(Qraw);
    const __bf16* __restrict__ Kg  = reinterpret_cast<const __bf16*>(Kraw);
    const __bf16* __restrict__ Vtg = reinterpret_cast<const __bf16*>(Vtraw);

    __shared__ __attribute__((aligned(16))) __bf16 kbuf[2][64 * 256]; // 2 x 32KB
    __shared__ __attribute__((aligned(16))) float  Sbuf[32 * 64];
    __shared__ __attribute__((aligned(16))) __bf16 Pbuf[32 * 64];
    __shared__ float red[32 * 8];
    __shared__ float mrow[32], lrow[32], arow[32];

    const int tid  = threadIdx.x;
    const int lane = tid & 31;
    const int w    = tid >> 5;
    const int b    = blockIdx.y;
    const int qbase = blockIdx.x * 32;

    const __bf16* Qb = Qg  + (size_t)b * Tt * Dd;
    const __bf16* Kb = Kg  + (size_t)b * Tt * Dd;
    const __bf16* Vb = Vtg + (size_t)b * Dd * Tt;

    const int qt = w >> 2;     // 0..1   (S phase)
    const int kt = w & 3;      // 0..3   (S phase)
    const int d0 = w * 128;    // PV-phase D slice

    const int ln15 = lane & 15;
    const int kbA  = (lane >> 4) * 8;
    const int kbB  = (lane >> 4) * 16;
    const int mhi  = (lane >> 4) * 8;

    const int prow = tid >> 3;         // softmax: row 0..31
    const int pcid = tid & 7;          // softmax: 8 threads per row

    const f32x8 z8 = {0.f, 0.f, 0.f, 0.f, 0.f, 0.f, 0.f, 0.f};
    f32x8 O[2][8];
#pragma unroll
    for (int i = 0; i < 2; ++i)
#pragma unroll
        for (int j = 0; j < 8; ++j) O[i][j] = z8;

    if (tid < 32) { mrow[tid] = -INFINITY; lrow[tid] = 0.f; }
    __syncthreads();

    const int nkt = (qbase + 32 + 63) >> 6;   // causal: key tiles 0..nkt-1
    const int total_chunks = nkt * 4;

#if HAVE_TDM
    if (w == 0)   // prime the pipeline: chunk 0 -> kbuf[0]
        tdm_load_k_chunk(Kb, (unsigned)(size_t)&kbuf[0][0]);
#endif

    for (int ktile = 0; ktile < nkt; ++ktile) {
        const int kt0 = ktile * 64;

        // ---------------- S = (Q K^T) tile for this wave ----------------
        f32x8 s = z8;
        const __bf16* qrow = Qb + (size_t)(qbase + qt * 16 + ln15) * Dd;

        for (int c = 0; c < 4; ++c) {
            const int gi = ktile * 4 + c;
#if HAVE_TDM
            if (w == 0) {
                if (gi + 1 < total_chunks) {
                    int nt_ = (gi + 1) >> 2, nc = (gi + 1) & 3;
                    tdm_load_k_chunk(Kb + (size_t)(nt_ * 64) * Dd + nc * 256,
                                     (unsigned)(size_t)&kbuf[(gi + 1) & 1][0]);
                    __builtin_amdgcn_s_wait_tensorcnt(1);  // chunk gi landed
                } else {
                    __builtin_amdgcn_s_wait_tensorcnt(0);
                }
            }
            __syncthreads();
#else
            { // cooperative synchronous copy of chunk gi
                const int row = tid >> 2, seg = tid & 3;
                const __bf16* src = Kb + (size_t)(kt0 + row) * Dd + c * 256 + seg * 64;
                __bf16* dst = &kbuf[gi & 1][row * 256 + seg * 64];
#pragma unroll
                for (int u = 0; u < 8; ++u)
                    *(bf16x8*)(dst + u * 8) = *(const bf16x8*)(src + u * 8);
            }
            __syncthreads();
#endif
            const __bf16* kbp = &kbuf[gi & 1][(kt * 16 + ln15) * 256];
#pragma unroll
            for (int dl = 0; dl < 8; ++dl) {
                const int dloc = dl * 32;
                const int dglo = c * 256 + dloc;
                BFrag a, bk;
                a.h[0]  = *(const bf16x8*)(qrow + dglo + kbA);
                a.h[1]  = *(const bf16x8*)(qrow + dglo + kbA + 16);
                bk.h[0] = *(const bf16x8*)(kbp + dloc + kbA);
                bk.h[1] = *(const bf16x8*)(kbp + dloc + kbA + 16);
                s = WMMA_BF16(a.v, bk.v, s);
            }
            __syncthreads();   // buffer reused two chunks later
        }

        // scale + causal mask + spill to LDS
#pragma unroll
        for (int r = 0; r < 8; ++r) {
            int mm = mhi + r;
            int gq = qbase + qt * 16 + mm;
            int gk = kt0 + kt * 16 + ln15;
            float sv = s[r] * 0.03125f;          // 1/sqrt(1024)
            if (gk > gq) sv = -INFINITY;
            Sbuf[(qt * 16 + mm) * 64 + kt * 16 + ln15] = sv;
        }
        __syncthreads();

        // ---------------- parallel online softmax ----------------
        {   // stage 1: per-thread partial max over 8 columns
            float pm = -INFINITY;
#pragma unroll
            for (int u = 0; u < 8; ++u)
                pm = fmaxf(pm, Sbuf[prow * 64 + pcid * 8 + u]);
            red[prow * 8 + pcid] = pm;
        }
        __syncthreads();
        if (tid < 32) {   // stage 2: row max, alpha
            float mold = mrow[tid];
            float mx = mold;
#pragma unroll
            for (int u = 0; u < 8; ++u) mx = fmaxf(mx, red[tid * 8 + u]);
            mrow[tid] = mx;
            arow[tid] = __expf(mold - mx);
        }
        __syncthreads();
        {   // stage 3: exp, write P (bf16), partial row sum
            float mx = mrow[prow];
            float ps = 0.f;
#pragma unroll
            for (int u = 0; u < 8; ++u) {
                float p = __expf(Sbuf[prow * 64 + pcid * 8 + u] - mx);
                Pbuf[prow * 64 + pcid * 8 + u] = (__bf16)p;
                ps += p;
            }
            red[prow * 8 + pcid] = ps;
        }
        __syncthreads();
        if (tid < 32) {   // stage 4: running row sum
            float l = lrow[tid] * arow[tid];
#pragma unroll
            for (int u = 0; u < 8; ++u) l += red[tid * 8 + u];
            lrow[tid] = l;
        }

        // rescale running O by alpha
#pragma unroll
        for (int i = 0; i < 2; ++i) {
#pragma unroll
            for (int r = 0; r < 8; ++r) {
                float al = arow[i * 16 + mhi + r];
#pragma unroll
                for (int j = 0; j < 8; ++j) O[i][j][r] *= al;
            }
        }

        // ---------------- O += P @ V  (this wave's 128-wide D slice) -----
#pragma unroll
        for (int ks = 0; ks < 2; ++ks) {
            BFrag a2[2];
#pragma unroll
            for (int i = 0; i < 2; ++i) {
                const __bf16* pr = &Pbuf[(i * 16 + ln15) * 64 + ks * 32];
                a2[i].h[0] = *(const bf16x8*)(pr + kbA);
                a2[i].h[1] = *(const bf16x8*)(pr + kbA + 16);
            }
#pragma unroll
            for (int j = 0; j < 8; ++j) {
                BFrag bv;
                const __bf16* vr = Vb + (size_t)(d0 + j * 16 + ln15) * Tt
                                      + kt0 + ks * 32;
                bv.h[0] = *(const bf16x8*)(vr + kbA);
                bv.h[1] = *(const bf16x8*)(vr + kbA + 16);
#pragma unroll
                for (int i = 0; i < 2; ++i)
                    O[i][j] = WMMA_BF16(a2[i].v, bv.v, O[i][j]);
            }
        }
        __syncthreads();   // Sbuf/Pbuf reused next iteration
    }

    // ---------------- epilogue: O / l, fp32 store ----------------
#pragma unroll
    for (int i = 0; i < 2; ++i) {
#pragma unroll
        for (int r = 0; r < 8; ++r) {
            int gq = qbase + i * 16 + mhi + r;
            float inv = 1.0f / lrow[i * 16 + mhi + r];
#pragma unroll
            for (int j = 0; j < 8; ++j) {
                int gd = d0 + j * 16 + ln15;
                Out[((size_t)b * Tt + gq) * Dd + gd] = O[i][j][r] * inv;
            }
        }
    }
}

// ---------------------------------------------------------------------------
extern "C" void kernel_launch(void* const* d_in, const int* in_sizes, int n_in,
                              void* d_out, int out_size, void* d_ws, size_t ws_size,
                              hipStream_t stream)
{
    const float* X  = (const float*)d_in[0];
    const float* Wq = (const float*)d_in[1];
    const float* Wk = (const float*)d_in[2];
    const float* Wv = (const float*)d_in[3];

    unsigned short* Qw = (unsigned short*)d_ws;                 // bf16 [B*T, D]
    unsigned short* Kw = Qw + (size_t)Mr * Dd;                  // bf16 [B*T, D]
    unsigned short* Vw = Kw + (size_t)Mr * Dd;                  // bf16 [B, D, T]

    dim3 gg(Dd / 64, Mr / 128);   // (16, 64)
    qkv_gemm_bf16<<<gg, 256, 0, stream>>>(X, Wq, Qw, 0);
    qkv_gemm_bf16<<<gg, 256, 0, stream>>>(X, Wk, Kw, 0);
    qkv_gemm_bf16<<<gg, 256, 0, stream>>>(X, Wv, Vw, 1);

    dim3 ga(Tt / 32, Bb);         // (64, 4)
    flash_attn_bf16<<<ga, 256, 0, stream>>>(Qw, Kw, Vw, (float*)d_out);
}
